// CrossGAT_80307298501084
// MI455X (gfx1250) — compile-verified, matrix operand order checked by reading, not run on hardware
//
#include <hip/hip_runtime.h>

typedef __bf16 bf16;
typedef bf16 bf16x8 __attribute__((ext_vector_type(8)));
typedef bf16 v16bf __attribute__((ext_vector_type(16)));
typedef float v8f __attribute__((ext_vector_type(8)));

#define DEVI __device__ __forceinline__

// ------------------------- constants -------------------------
static const int NN   = 20000;
static const int EIA  = 256000;
static const int EIR  = 256000;
static const int EIT  = 100000;
static const int ECH  = 32000;   // edge chunk (multiple of 16, divides EIA)
static const int ACH  = 50000;   // aux chunk  (multiple of 16, divides EIT)
static const int MAXK8 = 552;    // max Kpad (544) + 8 LDS anti-conflict pad

// ------------------------- device helpers -------------------------
DEVI float actf(float v, int act) {
  if (act == 1) return fmaxf(v, 0.f);
  if (act == 2) return v / (1.f + __expf(-v));   // silu
  return v;
}

DEVI unsigned enc_ord(float f) {   // order-preserving float->uint
  unsigned u = __float_as_uint(f);
  return (u & 0x80000000u) ? ~u : (u | 0x80000000u);
}

DEVI v16bf load_frag(const bf16* p) {
  // lane L holds row (L&15); chunks at k+[0..7] and k+[16..23], lanes>=16 shifted by 8
  bf16x8 lo = *reinterpret_cast<const bf16x8*>(p);
  bf16x8 hi = *reinterpret_cast<const bf16x8*>(p + 16);
  v16bf f;
#pragma unroll
  for (int t = 0; t < 8; ++t) { f[t] = lo[t]; f[t + 8] = hi[t]; }
  return f;
}

DEVI v16bf load_frag_s(const bf16* p) {   // LDS flavor (address space inferred -> ds_load)
  bf16x8 lo = *reinterpret_cast<const bf16x8*>(p);
  bf16x8 hi = *reinterpret_cast<const bf16x8*>(p + 16);
  v16bf f;
#pragma unroll
  for (int t = 0; t < 8; ++t) { f[t] = lo[t]; f[t + 8] = hi[t]; }
  return f;
}

// ------------------------- WMMA GEMM -------------------------
// C = act(A @ B + bias). A: M x K (bf16, lda=Kpad, K mult of 32, M mult of 16)
// Bt: Npad x K (bf16, pre-transposed, rows padded/zeroed to cover nt*16)
// Block = 8 waves covering 128 rows x one 32-col n-pair; the 32-row B panel is
// cooperatively staged in LDS once (conflict-free stride Kpad+8) and shared by
// all 8 waves; A streams from global with WGP-scope prefetch.
// Writes f32 (ldcf) and/or bf16 (ldcb); columns in [Ncols, ld) are written 0
// so K-pad regions of downstream GEMM inputs are zero-filled automatically.
__global__ void __launch_bounds__(256)
k_gemm(const bf16* __restrict__ A, int lda,
       const bf16* __restrict__ Bt, int ldb,
       const float* __restrict__ bias,
       float* __restrict__ Cf, int ldcf,
       bf16* __restrict__ Cb, int ldcb,
       int Mrows, int Ncols, int K, int mt, int nt, int act) {
  __shared__ __attribute__((aligned(16))) bf16 Bs[32 * MAXK8];
  int nt2 = (nt + 1) >> 1;
  int npair = blockIdx.x % nt2;
  int bm = blockIdx.x / nt2;
  int tn0 = npair * 2;
  int KP8 = K + 8;

  // cooperative stage of the 32-row B panel into LDS
  int kch = K >> 3;            // 16B chunks per row
  int chunks = 32 * kch;
  for (int c = threadIdx.x; c < chunks; c += 256) {
    int row = c / kch;
    int kc = (c - row * kch) << 3;
    int grow = tn0 * 16 + row;
    if (grow >= nt * 16) grow = nt * 16 - 1;   // safety for odd nt
    *reinterpret_cast<bf16x8*>(&Bs[row * KP8 + kc]) =
        *reinterpret_cast<const bf16x8*>(Bt + (size_t)grow * ldb + kc);
  }
  __syncthreads();

  int wave = threadIdx.x >> 5;
  int tmv = bm * 8 + wave;
  bool live = (tmv < mt);
  int tm = live ? tmv : (mt - 1);        // clamp: dead waves do harmless work
  int lane = threadIdx.x & 31;
  int fr   = lane & 15;
  int koff = (lane >> 4) * 8;

  const bf16* ap  = A + (size_t)(tm * 16 + fr) * lda + koff;
  const bf16* bs0 = &Bs[fr * KP8 + koff];
  const bf16* bs1 = &Bs[(16 + fr) * KP8 + koff];

  v8f accs[2];
  accs[0] = (v8f){0.f,0.f,0.f,0.f,0.f,0.f,0.f,0.f};
  accs[1] = (v8f){0.f,0.f,0.f,0.f,0.f,0.f,0.f,0.f};
  for (int k0 = 0; k0 < K; k0 += 32) {
    __builtin_prefetch(ap + k0 + 512, 0, 3);   // WGP-scope A stream-ahead prefetch
    v16bf a  = load_frag(ap + k0);
    v16bf b0 = load_frag_s(bs0 + k0);
    accs[0] = __builtin_amdgcn_wmma_f32_16x16x32_bf16(false, a, false, b0, (short)0, accs[0], false, false);
    v16bf b1 = load_frag_s(bs1 + k0);
    accs[1] = __builtin_amdgcn_wmma_f32_16x16x32_bf16(false, a, false, b1, (short)0, accs[1], false, false);
  }
  if (!live) return;

  int col = lane & 15, rbase = (lane >> 4) * 8;
#pragma unroll
  for (int t = 0; t < 2; ++t) {
    int n = (tn0 + t) * 16 + col;
    float bv = (bias && n < Ncols) ? bias[n] : 0.f;
#pragma unroll
    for (int r = 0; r < 8; ++r) {
      int m = tm * 16 + rbase + r;
      if (m >= Mrows) continue;
      float v = (n < Ncols) ? actf(accs[t][r] + bv, act) : 0.f;
      if (Cf && n < ldcf) Cf[(size_t)m * ldcf + n] = v;
      if (Cb && n < ldcb) Cb[(size_t)m * ldcb + n] = (bf16)v;
    }
  }
}

// ------------------------- small kernels -------------------------
__global__ void k_fill_u32(unsigned* __restrict__ p, unsigned v, long long n) {
  long long i = (long long)blockIdx.x * blockDim.x + threadIdx.x;
  if (i < n) p[i] = v;
}

__global__ void k_weight_t(const float* __restrict__ W, bf16* __restrict__ Bt,
                           int K, int Nc, int Kpad, int NcPad) {
  long long i = (long long)blockIdx.x * blockDim.x + threadIdx.x;
  if (i >= (long long)NcPad * Kpad) return;
  int n = (int)(i / Kpad), k = (int)(i % Kpad);
  float v = (n < Nc && k < K) ? W[(size_t)k * Nc + n] : 0.f;
  Bt[i] = (bf16)v;
}

__global__ void k_pack_bf16(const float* __restrict__ X, int ldi,
                            bf16* __restrict__ Y, int ldo, long long rows, int C) {
  long long i = (long long)blockIdx.x * blockDim.x + threadIdx.x;
  if (i >= rows * ldo) return;
  long long r = i / ldo; int c = (int)(i % ldo);
  Y[i] = (bf16)((c < C) ? X[r * ldi + c] : 0.f);
}

template <typename T>
__global__ void k_matvec(const T* __restrict__ A, int lda, const float* __restrict__ w,
                         const float* __restrict__ bias, float* __restrict__ out,
                         long long M, int K, int act) {
  long long r = (long long)blockIdx.x * blockDim.x + threadIdx.x;
  if (r >= M) return;
  const T* a = A + r * (long long)lda;
  float s = 0.f;
  for (int k = 0; k < K; ++k) s += (float)a[k] * w[k];
  if (bias) s += bias[0];
  if (act == 1) s = fmaxf(s, 0.f);
  out[r] = s;
}

// ---- GAT edge kernels ----
__global__ void k_edge_e(const int* __restrict__ ej, const int* __restrict__ ei,
                         int nE, int nTot, const float* __restrict__ es,
                         const float* __restrict__ ed, float* __restrict__ e,
                         unsigned* __restrict__ mxenc) {
  int t = blockIdx.x * blockDim.x + threadIdx.x;
  if (t >= nTot) return;
  int j, i;
  if (t < nE) { j = ej[t]; i = ei[t]; } else { j = i = t - nE; }
  float v = es[j] + ed[i];
  v = (v > 0.f) ? v : 0.2f * v;                // leaky_relu 0.2
  e[t] = v;
  atomicMax(&mxenc[i], enc_ord(v));
}

__global__ void k_mx_dec(const unsigned* __restrict__ enc, float* __restrict__ mx, int n) {
  int t = blockIdx.x * blockDim.x + threadIdx.x;
  if (t >= n) return;
  unsigned u = enc[t];
  unsigned v = (u & 0x80000000u) ? (u & 0x7FFFFFFFu) : ~u;
  float f = __uint_as_float(v);
  mx[t] = ((__float_as_uint(f) & 0x7F800000u) != 0x7F800000u) ? f : 0.f;
}

__global__ void k_edge_exp(const int* __restrict__ ei, int nE, int nTot,
                           float* __restrict__ e, const float* __restrict__ mx,
                           float* __restrict__ den) {
  int t = blockIdx.x * blockDim.x + threadIdx.x;
  if (t >= nTot) return;
  int i = (t < nE) ? ei[t] : (t - nE);
  float ex = __expf(e[t] - mx[i]);
  e[t] = ex;
  atomicAdd(&den[i], ex);
}

__global__ void k_gat_agg(const int* __restrict__ ej, const int* __restrict__ ei,
                          int nE, long long nTot, const float* __restrict__ ex,
                          const float* __restrict__ den, const float* __restrict__ hs,
                          float* __restrict__ out) {
  long long t = (long long)blockIdx.x * blockDim.x + threadIdx.x;
  if (t >= nTot * 128) return;
  long long eix = t >> 7; int f = (int)(t & 127);
  int j, i;
  if (eix < nE) { j = ej[eix]; i = ei[eix]; } else { j = i = (int)(eix - nE); }
  float alpha = ex[eix] / (den[i] + 1e-16f);
  atomicAdd(&out[(size_t)i * 128 + f], alpha * hs[(size_t)j * 128 + f]);
}

__global__ void k_bias_relu_pack(const float* __restrict__ acc, const float* __restrict__ b,
                                 float* __restrict__ xf, bf16* __restrict__ xb,
                                 long long rows, int C) {
  long long i = (long long)blockIdx.x * blockDim.x + threadIdx.x;
  if (i >= rows * C) return;
  int c = (int)(i % C);
  float v = fmaxf(acc[i] + b[c], 0.f);
  xf[i] = v; xb[i] = (bf16)v;
}

// ---- EGNN kernels ----
__global__ void k_edge_geom(const int* __restrict__ ej, const int* __restrict__ ei,
                            int c0, int ce, const float* __restrict__ coors,
                            float* __restrict__ rel, float* __restrict__ rd) {
  int t = blockIdx.x * blockDim.x + threadIdx.x;
  if (t >= ce) return;
  int j = ej[c0 + t], i = ei[c0 + t];
  float r0 = coors[(size_t)j*3+0] - coors[(size_t)i*3+0];
  float r1 = coors[(size_t)j*3+1] - coors[(size_t)i*3+1];
  float r2 = coors[(size_t)j*3+2] - coors[(size_t)i*3+2];
  rel[(size_t)t*3+0] = r0; rel[(size_t)t*3+1] = r1; rel[(size_t)t*3+2] = r2;
  rd[t] = r0*r0 + r1*r1 + r2*r2;
}

__global__ void k_egnn_pack(const int* __restrict__ ej, const int* __restrict__ ei,
                            int c0, int ce, const bf16* __restrict__ fb,
                            const float* __restrict__ rd, bf16* __restrict__ yo) {
  long long t = (long long)blockIdx.x * blockDim.x + threadIdx.x;
  if (t >= (long long)ce * 288) return;
  int e = (int)(t / 288), c = (int)(t % 288);
  int j = ej[c0 + e], i = ei[c0 + e];
  bf16 v;
  if (c < 128)       v = fb[(size_t)i * 128 + c];
  else if (c < 256)  v = fb[(size_t)j * 128 + (c - 128)];
  else if (c == 256) v = (bf16)rd[e];
  else               v = (bf16)0.f;
  yo[t] = v;
}

__global__ void k_scatter_m(const int* __restrict__ ei, int c0, int ce,
                            const float* __restrict__ mf, float* __restrict__ mi) {
  long long t = (long long)blockIdx.x * blockDim.x + threadIdx.x;
  if (t >= (long long)ce * 16) return;
  int e = (int)(t >> 4), f = (int)(t & 15);
  atomicAdd(&mi[(size_t)ei[c0 + e] * 16 + f], mf[t]);
}

__global__ void k_scatter_coor(const int* __restrict__ ei, int c0, int ce,
                               const float* __restrict__ cw, const float* __restrict__ rel,
                               float* __restrict__ co) {
  long long t = (long long)blockIdx.x * blockDim.x + threadIdx.x;
  if (t >= (long long)ce * 3) return;
  int e = (int)(t / 3), c = (int)(t % 3);
  atomicAdd(&co[(size_t)ei[c0 + e] * 3 + c], cw[e] * rel[(size_t)e*3 + c]);
}

__global__ void k_node_pack(const bf16* __restrict__ fb, const float* __restrict__ mi,
                            bf16* __restrict__ yo, long long rows) {
  long long t = (long long)blockIdx.x * blockDim.x + threadIdx.x;
  if (t >= rows * 160) return;
  long long r = t / 160; int c = (int)(t % 160);
  bf16 v;
  if (c < 128)      v = fb[r * 128 + c];
  else if (c < 144) v = (bf16)mi[r * 16 + (c - 128)];
  else              v = (bf16)0.f;
  yo[t] = v;
}

__global__ void k_residual_pack(const float* __restrict__ fin, const float* __restrict__ dlt,
                                float* __restrict__ fof, bf16* __restrict__ fob, long long n) {
  long long t = (long long)blockIdx.x * blockDim.x + threadIdx.x;
  if (t >= n) return;
  float v = fin[t] + dlt[t];
  fof[t] = v; fob[t] = (bf16)v;
}

__global__ void k_aux_pack(const int* __restrict__ e0, const int* __restrict__ e1,
                           int c0, int ce, const bf16* __restrict__ xa,
                           const bf16* __restrict__ xb, bf16* __restrict__ yo) {
  long long t = (long long)blockIdx.x * blockDim.x + threadIdx.x;
  if (t >= (long long)ce * 256) return;
  int r = (int)(t >> 8), c = (int)(t & 255);
  bf16 v;
  if (c < 128) v = xa[(size_t)e0[c0 + r] * 128 + c];
  else         v = xb[(size_t)e1[c0 + r] * 128 + (c - 128)];
  yo[t] = v;
}

// ------------------------- host helpers -------------------------
static inline int gblk(long long n) { return (int)((n + 255) / 256); }

static void gemm(hipStream_t st, const bf16* A, int lda, const bf16* Bt, int ldb,
                 const float* bias, float* Cf, int ldcf, bf16* Cb, int ldcb,
                 int M, int Ncols, int K, int act) {
  int ldmax = Ncols;
  if (Cf && ldcf > ldmax) ldmax = ldcf;
  if (Cb && ldcb > ldmax) ldmax = ldcb;
  int mt = (M + 15) / 16, nt = (ldmax + 15) / 16;
  int nt2 = (nt + 1) / 2;
  int mblks = (mt + 7) / 8;
  k_gemm<<<mblks * nt2, 256, 0, st>>>(A, lda, Bt, ldb, bias, Cf, ldcf, Cb, ldcb,
                                      M, Ncols, K, mt, nt, act);
}

static void wt(hipStream_t st, const float* W, bf16* Bt, int K, int Nc, int Kpad, int NcPad) {
  k_weight_t<<<gblk((long long)NcPad * Kpad), 256, 0, st>>>(W, Bt, K, Nc, Kpad, NcPad);
}

struct GatScratch {
  float *hs, *hd, *es, *ed, *ebuf, *mx, *den, *agg;
  unsigned* mxenc;
};

static void run_gat(hipStream_t st, const GatScratch& g,
                    const bf16* srcb, const bf16* dstb, int Kpad,
                    const bf16* Wst, const bf16* Wdt,
                    const float* a_s, const float* a_d, const float* bvec,
                    const int* ej, const int* ei, int nE, int nself,
                    int n_src, int n_dst, float* outf, bf16* outb) {
  gemm(st, srcb, Kpad, Wst, Kpad, nullptr, g.hs, 128, nullptr, 0, n_src, 128, Kpad, 0);
  gemm(st, dstb, Kpad, Wdt, Kpad, nullptr, g.hd, 128, nullptr, 0, n_dst, 128, Kpad, 0);
  k_matvec<float><<<gblk(n_src), 256, 0, st>>>(g.hs, 128, a_s, nullptr, g.es, n_src, 128, 0);
  k_matvec<float><<<gblk(n_dst), 256, 0, st>>>(g.hd, 128, a_d, nullptr, g.ed, n_dst, 128, 0);
  k_fill_u32<<<gblk(n_dst), 256, 0, st>>>(g.mxenc, 0x007FFFFFu, n_dst);      // enc(-inf)
  k_fill_u32<<<gblk(n_dst), 256, 0, st>>>((unsigned*)g.den, 0u, n_dst);
  k_fill_u32<<<gblk((long long)n_dst * 128), 256, 0, st>>>((unsigned*)g.agg, 0u, (long long)n_dst * 128);
  int nTot = nE + nself;
  k_edge_e<<<gblk(nTot), 256, 0, st>>>(ej, ei, nE, nTot, g.es, g.ed, g.ebuf, g.mxenc);
  k_mx_dec<<<gblk(n_dst), 256, 0, st>>>(g.mxenc, g.mx, n_dst);
  k_edge_exp<<<gblk(nTot), 256, 0, st>>>(ei, nE, nTot, g.ebuf, g.mx, g.den);
  k_gat_agg<<<gblk((long long)nTot * 128), 256, 0, st>>>(ej, ei, nE, (long long)nTot,
                                                         g.ebuf, g.den, g.hs, g.agg);
  k_bias_relu_pack<<<gblk((long long)n_dst * 128), 256, 0, st>>>(g.agg, bvec, outf, outb,
                                                                 (long long)n_dst, 128);
}

struct EgnnW {
  const bf16 *ew1t, *ew2t, *cw1t, *nw1t, *nw2t;
  const float *eb1, *eb2, *cb1, *cw2, *cb2, *nb1, *nb2;
};
struct EgnnScratch {
  bf16 *minb, *hidb, *mb, *cwhb, *npack, *nh;
  float *mf, *cwf, *rel, *rd, *m_i, *nout;
};

static void run_egnn(hipStream_t st, const EgnnW& w, const EgnnScratch& s,
                     const float* coors_in, float* coors_out,
                     const float* fin_f, const bf16* fin_b,
                     float* fout_f, bf16* fout_b,
                     const int* ej, const int* ei, int nE, int nn) {
  hipMemcpyAsync(coors_out, coors_in, (size_t)nn * 3 * sizeof(float),
                 hipMemcpyDeviceToDevice, st);
  k_fill_u32<<<gblk((long long)nn * 16), 256, 0, st>>>((unsigned*)s.m_i, 0u, (long long)nn * 16);
  for (int c0 = 0; c0 < nE; c0 += ECH) {
    int ce = (nE - c0 < ECH) ? (nE - c0) : ECH;
    k_edge_geom<<<gblk(ce), 256, 0, st>>>(ej, ei, c0, ce, coors_in, s.rel, s.rd);
    k_egnn_pack<<<gblk((long long)ce * 288), 256, 0, st>>>(ej, ei, c0, ce, fin_b, s.rd, s.minb);
    gemm(st, s.minb, 288, w.ew1t, 288, w.eb1, nullptr, 0, s.hidb, 544, ce, 514, 288, 2);
    gemm(st, s.hidb, 544, w.ew2t, 544, w.eb2, s.mf, 16, s.mb, 32, ce, 16, 544, 2);
    gemm(st, s.mb, 32, w.cw1t, 32, w.cb1, nullptr, 0, s.cwhb, 64, ce, 64, 32, 2);
    k_matvec<bf16><<<gblk(ce), 256, 0, st>>>(s.cwhb, 64, w.cw2, w.cb2, s.cwf, ce, 64, 0);
    k_scatter_m<<<gblk((long long)ce * 16), 256, 0, st>>>(ei, c0, ce, s.mf, s.m_i);
    k_scatter_coor<<<gblk((long long)ce * 3), 256, 0, st>>>(ei, c0, ce, s.cwf, s.rel, coors_out);
  }
  k_node_pack<<<gblk((long long)nn * 160), 256, 0, st>>>(fin_b, s.m_i, s.npack, nn);
  gemm(st, s.npack, 160, w.nw1t, 160, w.nb1, nullptr, 0, s.nh, 256, nn, 256, 160, 2);
  gemm(st, s.nh, 256, w.nw2t, 256, w.nb2, s.nout, 128, nullptr, 0, nn, 128, 256, 0);
  k_residual_pack<<<gblk((long long)nn * 128), 256, 0, st>>>(fin_f, s.nout, fout_f, fout_b,
                                                             (long long)nn * 128);
}

// ------------------------- entry point -------------------------
extern "C" void kernel_launch(void* const* d_in, const int* in_sizes, int n_in,
                              void* d_out, int out_size, void* d_ws, size_t ws_size,
                              hipStream_t stream) {
  (void)in_sizes; (void)n_in; (void)out_size; (void)ws_size;
  const float* x1   = (const float*)d_in[0];
  const float* x2   = (const float*)d_in[1];
  const float* pos1 = (const float*)d_in[2];
  const float* pos2 = (const float*)d_in[3];
  const int* ei_i1  = (const int*)d_in[4];
  const int* ei_i2  = (const int*)d_in[5];
  const int* ei_12  = (const int*)d_in[6];
  const int* ei_21  = (const int*)d_in[7];
  const int* ei_int = (const int*)d_in[8];
  const float* g0_w  = (const float*)d_in[9];
  const float* g0_as = (const float*)d_in[10];
  const float* g0_ad = (const float*)d_in[11];
  const float* g0_b  = (const float*)d_in[12];
  const float* g1_w  = (const float*)d_in[13];
  const float* g1_as = (const float*)d_in[14];
  const float* g1_ad = (const float*)d_in[15];
  const float* g1_b  = (const float*)d_in[16];
  const float* it_ws = (const float*)d_in[17];
  const float* it_wd = (const float*)d_in[18];
  const float* it_as = (const float*)d_in[19];
  const float* it_ad = (const float*)d_in[20];
  const float* it_b  = (const float*)d_in[21];
  const float* eg_e_w1 = (const float*)d_in[22];
  const float* eg_e_b1 = (const float*)d_in[23];
  const float* eg_e_w2 = (const float*)d_in[24];
  const float* eg_e_b2 = (const float*)d_in[25];
  const float* eg_c_w1 = (const float*)d_in[26];
  const float* eg_c_b1 = (const float*)d_in[27];
  const float* eg_c_w2 = (const float*)d_in[28];
  const float* eg_c_b2 = (const float*)d_in[29];
  const float* eg_n_w1 = (const float*)d_in[30];
  const float* eg_n_b1 = (const float*)d_in[31];
  const float* eg_n_w2 = (const float*)d_in[32];
  const float* eg_n_b2 = (const float*)d_in[33];
  const float* lin_w  = (const float*)d_in[34];
  const float* lin_b  = (const float*)d_in[35];
  const float* aux_w1 = (const float*)d_in[36];
  const float* aux_b1 = (const float*)d_in[37];
  const float* aux_w2 = (const float*)d_in[38];
  const float* aux_b2 = (const float*)d_in[39];
  float* out = (float*)d_out;

  // bump allocator over workspace
  char* base = (char*)d_ws;
  size_t off = 0;
  auto alloc = [&](size_t bytes) -> void* {
    void* p = base + off;
    off += (bytes + 255) & ~(size_t)255;
    return p;
  };

  // ---- transpose/pack weights to bf16 (Npad x Kpad) ----
  bf16 *g0t[2], *g1t[2], *itst[2][2], *itdt[2][2];
  bf16 *ew1t[2][2], *ew2t[2][2], *cw1t[2][2], *nw1t[2][2], *nw2t[2][2], *aux1t;
  for (int g = 0; g < 2; ++g) {
    g0t[g] = (bf16*)alloc((size_t)128 * 64 * 2);
    wt(stream, g0_w + (size_t)g * 64 * 128, g0t[g], 64, 128, 64, 128);
    g1t[g] = (bf16*)alloc((size_t)128 * 128 * 2);
    wt(stream, g1_w + (size_t)g * 128 * 128, g1t[g], 128, 128, 128, 128);
  }
  for (int l = 0; l < 2; ++l)
    for (int d = 0; d < 2; ++d) {
      int s = l * 2 + d;
      itst[l][d] = (bf16*)alloc((size_t)128 * 128 * 2);
      wt(stream, it_ws + (size_t)s * 128 * 128, itst[l][d], 128, 128, 128, 128);
      itdt[l][d] = (bf16*)alloc((size_t)128 * 128 * 2);
      wt(stream, it_wd + (size_t)s * 128 * 128, itdt[l][d], 128, 128, 128, 128);
      ew1t[l][d] = (bf16*)alloc((size_t)544 * 288 * 2);
      wt(stream, eg_e_w1 + (size_t)s * 257 * 514, ew1t[l][d], 257, 514, 288, 544);
      ew2t[l][d] = (bf16*)alloc((size_t)32 * 544 * 2);
      wt(stream, eg_e_w2 + (size_t)s * 514 * 16, ew2t[l][d], 514, 16, 544, 32);
      cw1t[l][d] = (bf16*)alloc((size_t)64 * 32 * 2);
      wt(stream, eg_c_w1 + (size_t)s * 16 * 64, cw1t[l][d], 16, 64, 32, 64);
      nw1t[l][d] = (bf16*)alloc((size_t)256 * 160 * 2);
      wt(stream, eg_n_w1 + (size_t)s * 144 * 256, nw1t[l][d], 144, 256, 160, 256);
      nw2t[l][d] = (bf16*)alloc((size_t)128 * 256 * 2);
      wt(stream, eg_n_w2 + (size_t)s * 256 * 128, nw2t[l][d], 256, 128, 256, 128);
    }
  aux1t = (bf16*)alloc((size_t)128 * 256 * 2);
  wt(stream, aux_w1, aux1t, 256, 128, 256, 128);

  // ---- node buffers ----
  float *f1[2], *f2[2];
  bf16  *b1[2], *b2[2];
  for (int p = 0; p < 2; ++p) {
    f1[p] = (float*)alloc((size_t)NN * 128 * 4);
    f2[p] = (float*)alloc((size_t)NN * 128 * 4);
    b1[p] = (bf16*)alloc((size_t)NN * 128 * 2);
    b2[p] = (bf16*)alloc((size_t)NN * 128 * 2);
  }
  bf16* inb1 = (bf16*)alloc((size_t)NN * 64 * 2);
  bf16* inb2 = (bf16*)alloc((size_t)NN * 64 * 2);
  float* c1a = (float*)alloc((size_t)NN * 3 * 4);
  float* c1b = (float*)alloc((size_t)NN * 3 * 4);
  float* c2a = (float*)alloc((size_t)NN * 3 * 4);
  float* c2b = (float*)alloc((size_t)NN * 3 * 4);

  // ---- GAT scratch ----
  GatScratch gs;
  gs.hs    = (float*)alloc((size_t)NN * 128 * 4);
  gs.hd    = (float*)alloc((size_t)NN * 128 * 4);
  gs.es    = (float*)alloc((size_t)NN * 4);
  gs.ed    = (float*)alloc((size_t)NN * 4);
  gs.ebuf  = (float*)alloc((size_t)(EIA + NN) * 4);
  gs.mxenc = (unsigned*)alloc((size_t)NN * 4);
  gs.mx    = (float*)alloc((size_t)NN * 4);
  gs.den   = (float*)alloc((size_t)NN * 4);
  gs.agg   = (float*)alloc((size_t)NN * 128 * 4);

  // ---- EGNN scratch (chunked) ----
  EgnnScratch es_;
  es_.minb  = (bf16*)alloc((size_t)ECH * 288 * 2);
  es_.hidb  = (bf16*)alloc((size_t)ECH * 544 * 2);
  es_.mb    = (bf16*)alloc((size_t)ECH * 32 * 2);
  es_.cwhb  = (bf16*)alloc((size_t)ECH * 64 * 2);
  es_.mf    = (float*)alloc((size_t)ECH * 16 * 4);
  es_.cwf   = (float*)alloc((size_t)ECH * 4);
  es_.rel   = (float*)alloc((size_t)ECH * 3 * 4);
  es_.rd    = (float*)alloc((size_t)ECH * 4);
  es_.m_i   = (float*)alloc((size_t)NN * 16 * 4);
  es_.npack = (bf16*)alloc((size_t)NN * 160 * 2);
  es_.nh    = (bf16*)alloc((size_t)NN * 256 * 2);
  es_.nout  = (float*)alloc((size_t)NN * 128 * 4);

  // ---- aux scratch ----
  bf16* xib   = (bf16*)alloc((size_t)ACH * 256 * 2);
  bf16* auxhb = (bf16*)alloc((size_t)ACH * 128 * 2);

  // ---- pack inputs ----
  k_pack_bf16<<<gblk((long long)NN * 64), 256, 0, stream>>>(x1, 64, inb1, 64, NN, 64);
  k_pack_bf16<<<gblk((long long)NN * 64), 256, 0, stream>>>(x2, 64, inb2, 64, NN, 64);

  const int *ej1 = ei_i1, *eid1 = ei_i1 + EIA;
  const int *ej2 = ei_i2, *eid2 = ei_i2 + EIA;
  const int *ej12 = ei_12, *eid12 = ei_12 + EIR;
  const int *ej21 = ei_21, *eid21 = ei_21 + EIR;

  // ---- intra GAT layers (with self loops) ----
  run_gat(stream, gs, inb1, inb1, 64, g0t[0], g0t[0], g0_as, g0_ad, g0_b,
          ej1, eid1, EIA, NN, NN, NN, f1[0], b1[0]);
  run_gat(stream, gs, inb2, inb2, 64, g0t[1], g0t[1], g0_as + 128, g0_ad + 128, g0_b + 128,
          ej2, eid2, EIA, NN, NN, NN, f2[0], b2[0]);
  run_gat(stream, gs, b1[0], b1[0], 128, g1t[0], g1t[0], g1_as, g1_ad, g1_b,
          ej1, eid1, EIA, NN, NN, NN, f1[1], b1[1]);
  run_gat(stream, gs, b2[0], b2[0], 128, g1t[1], g1t[1], g1_as + 128, g1_ad + 128, g1_b + 128,
          ej2, eid2, EIA, NN, NN, NN, f2[1], b2[1]);

  // ---- inter GAT layers (no self loops) ----
  int c1i = 1, c2i = 1;
  for (int l = 0; l < 2; ++l) {
    run_gat(stream, gs, b1[c1i], b2[c2i], 128, itst[l][0], itdt[l][0],
            it_as + (size_t)(l * 2 + 0) * 128, it_ad + (size_t)(l * 2 + 0) * 128,
            it_b + (size_t)(l * 2 + 0) * 128,
            ej12, eid12, EIR, 0, NN, NN, f2[1 - c2i], b2[1 - c2i]);
    run_gat(stream, gs, b2[c2i], b1[c1i], 128, itst[l][1], itdt[l][1],
            it_as + (size_t)(l * 2 + 1) * 128, it_ad + (size_t)(l * 2 + 1) * 128,
            it_b + (size_t)(l * 2 + 1) * 128,
            ej21, eid21, EIR, 0, NN, NN, f1[1 - c1i], b1[1 - c1i]);
    c1i ^= 1; c2i ^= 1;
  }

  // ---- EGNN layers ----
  const float* curc1 = pos1;
  const float* curc2 = pos2;
  for (int l = 0; l < 2; ++l) {
    EgnnW w0 = { ew1t[l][0], ew2t[l][0], cw1t[l][0], nw1t[l][0], nw2t[l][0],
                 eg_e_b1 + (size_t)(l * 2 + 0) * 514, eg_e_b2 + (size_t)(l * 2 + 0) * 16,
                 eg_c_b1 + (size_t)(l * 2 + 0) * 64,  eg_c_w2 + (size_t)(l * 2 + 0) * 64,
                 eg_c_b2 + (size_t)(l * 2 + 0),       eg_n_b1 + (size_t)(l * 2 + 0) * 256,
                 eg_n_b2 + (size_t)(l * 2 + 0) * 128 };
    float* co1 = (l == 0) ? c1a : c1b;
    run_egnn(stream, w0, es_, curc1, co1, f1[c1i], b1[c1i], f1[1 - c1i], b1[1 - c1i],
             ej1, eid1, EIA, NN);
    curc1 = co1; c1i ^= 1;

    EgnnW w1 = { ew1t[l][1], ew2t[l][1], cw1t[l][1], nw1t[l][1], nw2t[l][1],
                 eg_e_b1 + (size_t)(l * 2 + 1) * 514, eg_e_b2 + (size_t)(l * 2 + 1) * 16,
                 eg_c_b1 + (size_t)(l * 2 + 1) * 64,  eg_c_w2 + (size_t)(l * 2 + 1) * 64,
                 eg_c_b2 + (size_t)(l * 2 + 1),       eg_n_b1 + (size_t)(l * 2 + 1) * 256,
                 eg_n_b2 + (size_t)(l * 2 + 1) * 128 };
    float* co2 = (l == 0) ? c2a : c2b;
    run_egnn(stream, w1, es_, curc2, co2, f2[c2i], b2[c2i], f2[1 - c2i], b2[1 - c2i],
             ej2, eid2, EIA, NN);
    curc2 = co2; c2i ^= 1;
  }

  // ---- outputs ----
  float* x1fin = f1[c1i]; bf16* x1bfin = b1[c1i];
  float* x2fin = f2[c2i]; bf16* x2bfin = b2[c2i];
  k_matvec<float><<<gblk(NN), 256, 0, stream>>>(x1fin, 128, lin_w, lin_b, out, NN, 128, 0);
  k_matvec<float><<<gblk(NN), 256, 0, stream>>>(x2fin, 128, lin_w, lin_b, out + NN, NN, 128, 0);

  const int* eiI0 = ei_int;
  const int* eiI1 = ei_int + EIT;
  for (int c0 = 0; c0 < EIT; c0 += ACH) {
    int ce = (EIT - c0 < ACH) ? (EIT - c0) : ACH;
    k_aux_pack<<<gblk((long long)ce * 256), 256, 0, stream>>>(eiI0, eiI1, c0, ce,
                                                              x1bfin, x2bfin, xib);
    gemm(stream, xib, 256, aux1t, 256, aux_b1, nullptr, 0, auxhb, 128, ce, 128, 256, 1);
    k_matvec<bf16><<<gblk(ce), 256, 0, stream>>>(auxhb, 128, aux_w2, aux_b2,
                                                 out + 2 * (size_t)NN + c0, ce, 128, 1);
  }
}
